// WaveNet_27169963114889
// MI455X (gfx1250) — matrix operand order repaired and checked by reading, not compiled, not run on hardware
//
#include <hip/hip_runtime.h>
#include <hip/hip_bf16.h>

// CDNA5 / gfx1250 WaveNet forward.
// WMMA (v_wmma_f32_16x16x32_bf16, wave32) register layouts per CDNA5 ISA 7.12.2:
//  A (16x32 bf16, v16bf): lane L -> row M=L&15; kbase=(L<16)?0:8;
//                         element e -> K = kbase + (e<8 ? e : e+8)
//  B (32x16 bf16, v16bf): lane L -> col N=L&15; khalf=L>>4;
//                         element e -> K = khalf*8 + (e<8 ? e : e+8)
//  C/D (16x16 f32, v8f):  lane L -> col N=L&15; element r -> row M = r + 8*(L>>4)
//
// Per-layer fused conv+tanh+res kernels store tanh output y as bf16 pre-swizzled
// into the per-lane WMMA B-register layout (32 contiguous bytes / lane / tile).
// The skip path is one fused kernel: y tiles are staged global->LDS with
// GLOBAL_LOAD_ASYNC_TO_LDS_B128 (double buffered, ASYNCcnt-tracked), then 28
// chained WMMAs per tile + bias + relu + in-register time reduction.
// skip_sum never touches HBM.

typedef __attribute__((ext_vector_type(16))) __bf16 v16bf;
typedef __attribute__((ext_vector_type(8)))  float  v8f;

#define TLEN 16384
#define NTILE (TLEN / 16)   // 1024 time tiles per (b)
#define BATCH 8
#define NLAYER 28

__device__ __constant__ int g_dil[NLAYER] = {1, 2, 4, 8, 16, 32, 64, 128, 256, 512,
                                                2, 4, 8, 16, 32, 64, 128, 256, 512,
                                                2, 4, 8, 16, 32, 64, 128, 256, 512};

static __device__ __forceinline__ v8f wmma_bf16(v16bf a, v16bf b, v8f c) {
  return __builtin_amdgcn_wmma_f32_16x16x32_bf16(false, a, false, b, (short)0, c,
                                                 false, false);
}

// Async DMA: copy 16B from global to LDS, tracked by ASYNCcnt (CDNA5 ISA 10.7).
static __device__ __forceinline__ void async_g2l_b128(unsigned lds_off,
                                                      const void* gaddr) {
  asm volatile("global_load_async_to_lds_b128 %0, %1, off"
               :: "v"(lds_off), "v"(gaddr)
               : "memory");
}
static __device__ __forceinline__ void wait_async0() {
  asm volatile("s_wait_asynccnt 0x0" ::: "memory");
}

// ---------------------------------------------------------------------------
// Pre-pack Ws[28][256][32] f32 -> A-layout bf16 tiles wspk[28][16][32][16],
// and zero the atomic accumulator racc[B][256].
// ---------------------------------------------------------------------------
__global__ __launch_bounds__(256) void wavenet_prepack_kernel(
    const float* __restrict__ Ws, float* __restrict__ racc,
    __bf16* __restrict__ wspk) {
  const int idx = blockIdx.x * 256 + threadIdx.x;
  if (idx < NLAYER * 16 * 32 * 16) {
    const int e = idx & 15;
    const int L = (idx >> 4) & 31;
    const int s = (idx >> 9) & 15;
    const int i = idx >> 13;
    const int M = s * 16 + (L & 15);
    const int kb = (L < 16) ? 0 : 8;
    const int K = kb + ((e < 8) ? e : (e + 8));
    wspk[idx] = (__bf16)Ws[((size_t)i * 256 + M) * 32 + K];
  }
  if (idx < BATCH * 256) racc[idx] = 0.f;
}

// ---------------------------------------------------------------------------
// One WaveNet layer: y = tanh(dilated conv2(x)); write y packed bf16 (B-layout,
// zeros in the invalid tail); xout = xin + Wr@y + br (br everywhere).
// Block: 128 threads = 4 waves; each wave owns 4 independent 16-wide time tiles.
// ---------------------------------------------------------------------------
__global__ __launch_bounds__(128) void wavenet_layer_kernel(
    const float* __restrict__ xin,   // [B,32,T] f32 ([B,1,T] if in1)
    float* __restrict__ xout,        // [B,32,T] f32
    __bf16* __restrict__ ypk,        // [28][B][NTILE][32][16] bf16
    const float* __restrict__ Wcw,   // [32,IC,2] f32
    const float* __restrict__ bcv,   // [32]
    const float* __restrict__ Wr,    // [32,32]
    const float* __restrict__ brv,   // [32]
    int d, int in1, int layer) {
  __shared__ __bf16 lds_y[4][32][18];  // per-wave y staging (C -> B relayout)
  __shared__ float lds_br[32];
  __shared__ float lds_bc[32];

  const int tid  = threadIdx.x;
  const int lane = tid & 31;
  const int wv   = tid >> 5;
  const int b    = blockIdx.y;
  const int wg_t0 = blockIdx.x * 256;

  if (tid < 32) { lds_br[tid] = brv[tid]; lds_bc[tid] = bcv[tid]; }
  __syncthreads();

  // per-lane A tiles for conv taps and residual matrix
  const int Lm    = lane & 15;
  const int kbase = (lane < 16) ? 0 : 8;
  v16bf a_c[2][2];  // [tap][o-tile]
  if (!in1) {
    for (int tap = 0; tap < 2; ++tap)
      for (int ot = 0; ot < 2; ++ot) {
        const int M = ot * 16 + Lm;
        v16bf a;
        for (int e = 0; e < 16; ++e) {
          const int K = kbase + ((e < 8) ? e : (e + 8));
          a[e] = (__bf16)Wcw[(M * 32 + K) * 2 + tap];
        }
        a_c[tap][ot] = a;
      }
  }
  v16bf a_r[2];
  for (int ot = 0; ot < 2; ++ot) {
    const int M = ot * 16 + Lm;
    v16bf a;
    for (int e = 0; e < 16; ++e) {
      const int K = kbase + ((e < 8) ? e : (e + 8));
      a[e] = (__bf16)Wr[M * 32 + K];
    }
    a_r[ot] = a;
  }

  const int n     = lane & 15;  // time column within tile
  const int khalf = lane >> 4;

  for (int j = 0; j < 4; ++j) {
    const int t0   = wg_t0 + (wv * 4 + j) * 16;
    const int tile = t0 >> 4;
    const int tcur = t0 + n;
    const bool valid = tcur < (TLEN - d);

    // ---- y = tanh(conv) in C layout ----
    v8f y0, y1;
    if (in1) {
      int tB = tcur + d; if (tB > TLEN - 1) tB = TLEN - 1;
      const float xa = xin[b * TLEN + tcur];
      const float xb = xin[b * TLEN + tB];
      for (int r = 0; r < 8; ++r) {
        const int m0 = r + 8 * khalf, m1 = m0 + 16;
        y0[r] = valid ? tanhf(Wcw[m0 * 2 + 0] * xa + Wcw[m0 * 2 + 1] * xb + lds_bc[m0]) : 0.f;
        y1[r] = valid ? tanhf(Wcw[m1 * 2 + 0] * xa + Wcw[m1 * 2 + 1] * xb + lds_bc[m1]) : 0.f;
      }
    } else {
      int tB = tcur + d; if (tB > TLEN - 1) tB = TLEN - 1;
      v16bf bx, bxd;  // B-layout activations for the two taps
      for (int e = 0; e < 16; ++e) {
        const int k = khalf * 8 + ((e < 8) ? e : (e + 8));
        bx[e]  = (__bf16)xin[(b * 32 + k) * TLEN + tcur];
        bxd[e] = (__bf16)xin[(b * 32 + k) * TLEN + tB];
      }
      v8f c0, c1;
      for (int r = 0; r < 8; ++r) { c0[r] = 0.f; c1[r] = 0.f; }
      c0 = wmma_bf16(a_c[0][0], bx, c0);
      c0 = wmma_bf16(a_c[1][0], bxd, c0);
      c1 = wmma_bf16(a_c[0][1], bx, c1);
      c1 = wmma_bf16(a_c[1][1], bxd, c1);
      for (int r = 0; r < 8; ++r) {
        const int m0 = r + 8 * khalf;
        y0[r] = valid ? tanhf(c0[r] + lds_bc[m0]) : 0.f;
        y1[r] = valid ? tanhf(c1[r] + lds_bc[m0 + 16]) : 0.f;
      }
    }

    // ---- C-layout -> B-layout via LDS ----
    for (int r = 0; r < 8; ++r) {
      lds_y[wv][r + 8 * khalf][n]      = (__bf16)y0[r];
      lds_y[wv][r + 8 * khalf + 16][n] = (__bf16)y1[r];
    }
    __syncthreads();
    v16bf by;
    for (int e = 0; e < 16; ++e) by[e] = lds_y[wv][khalf * 8 + ((e < 8) ? e : (e + 8))][n];

    // ---- publish y in packed per-lane B-layout (32B contiguous per lane) ----
    *(v16bf*)(ypk + ((((size_t)layer * BATCH + b) * NTILE + tile) * 32 + lane) * 16) = by;

    // ---- residual: xout = xin + Wr@y + br ----
    v8f r0, r1;
    for (int r = 0; r < 8; ++r) {
      const int m0 = r + 8 * khalf, m1 = m0 + 16;
      const float xo0 = in1 ? xin[b * TLEN + tcur] : xin[(b * 32 + m0) * TLEN + tcur];
      const float xo1 = in1 ? xo0 : xin[(b * 32 + m1) * TLEN + tcur];
      r0[r] = xo0 + lds_br[m0];
      r1[r] = xo1 + lds_br[m1];
    }
    r0 = wmma_bf16(a_r[0], by, r0);
    r1 = wmma_bf16(a_r[1], by, r1);
    for (int r = 0; r < 8; ++r) {
      const int m0 = r + 8 * khalf, m1 = m0 + 16;
      xout[(b * 32 + m0) * TLEN + tcur] = r0[r];
      xout[(b * 32 + m1) * TLEN + tcur] = r1[r];
    }
    __syncthreads();
  }
}

// ---------------------------------------------------------------------------
// Fused skip path. Block: 512 threads = 16 waves; wave w owns output-channel
// tile w. Per 16-wide time tile:
//   stage y (28 layers x 1KB) global->LDS via async DMA (double-buffered),
//   c = bias(t); c += Ws_i @ y_i (28 chained WMMAs); acc += relu(c);
// then reduce acc over the 16 time columns, atomicAdd into racc[B,256].
// ---------------------------------------------------------------------------
__global__ __launch_bounds__(512) void wavenet_skipsum_kernel(
    const __bf16* __restrict__ ypk,   // [28][B][NTILE][32][16]
    const __bf16* __restrict__ wspk,  // [28][16][32][16] A-layout
    const float* __restrict__ bs,     // [28][256]
    float* __restrict__ racc) {       // [B][256]
  __shared__ __attribute__((aligned(32))) __bf16 ybuf[2][NLAYER * 512];  // 2x28KB
  __shared__ float lds_bfull[256];
  __shared__ float lds_red[16][32][8];

  const int tid  = threadIdx.x;
  const int lane = tid & 31;
  const int wv   = tid >> 5;  // s-tile 0..15
  const int b    = blockIdx.y;
  const int tile0 = blockIdx.x * 32;

  for (int c = tid; c < 256; c += 512) {
    float s = 0.f;
    for (int i = 0; i < NLAYER; ++i) s += bs[i * 256 + c];
    lds_bfull[c] = s;
  }

  // Issue async global->LDS copies of one tile's y (28 x 1KB chunks).
  auto issue_tile = [&](int tile, int bufsel) {
    for (int k = tid; k < NLAYER * 64; k += 512) {
      const int i   = k >> 6;   // layer
      const int off = k & 63;   // 16B segment within the 1KB chunk
      const __bf16* g = ypk +
          ((((size_t)i * BATCH + b) * NTILE + tile) * 32) * 16 + (size_t)off * 8;
      const unsigned l =
          (unsigned)(uintptr_t)&ybuf[bufsel][i * 512 + off * 8];
      async_g2l_b128(l, (const void*)g);
    }
  };

  issue_tile(tile0, 0);
  wait_async0();
  __syncthreads();  // async data + lds_bfull visible to all waves

  const int n     = lane & 15;
  const int khalf = lane >> 4;

  float acc[8];
  for (int r = 0; r < 8; ++r) acc[r] = 0.f;

  for (int j = 0; j < 32; ++j) {
    const int tile  = tile0 + j;
    const int tbase = tile * 16;
    const int tcur  = tbase + n;
    const int cur   = j & 1;

    // prefetch next tile into the other buffer (safe: its last readers passed
    // the barrier at the end of iteration j-1)
    if (j + 1 < 32) issue_tile(tile + 1, cur ^ 1);

    v8f c;
    if (tbase + 15 < TLEN - 512) {  // all 28 layers valid across this tile
      for (int r = 0; r < 8; ++r) c[r] = lds_bfull[wv * 16 + r + 8 * khalf];
    } else {                         // tail: per-lane masked bias sum (rare)
      for (int r = 0; r < 8; ++r) {
        const int m = wv * 16 + r + 8 * khalf;
        float s = 0.f;
        for (int i = 0; i < NLAYER; ++i)
          if (tcur < TLEN - g_dil[i]) s += bs[i * 256 + m];
        c[r] = s;
      }
    }

    for (int i = 0; i < NLAYER; ++i) {
      const v16bf aw = *(const v16bf*)(wspk + (((size_t)i * 16 + wv) * 32 + lane) * 16);
      const v16bf yv = *(const v16bf*)&ybuf[cur][i * 512 + lane * 16];
      c = wmma_bf16(aw, yv, c);
    }
    for (int r = 0; r < 8; ++r) acc[r] += fmaxf(c[r], 0.f);

    wait_async0();    // prefetch landed
    __syncthreads();  // all waves done reading `cur`, next buffer visible
  }

  // reduce over the 16 time columns held across lanes
  for (int r = 0; r < 8; ++r) lds_red[wv][lane][r] = acc[r];
  __syncthreads();
  if (lane < 16) {
    const int m  = lane;          // row within the wave's 16-channel tile
    const int kh = m >> 3;
    const int r  = m & 7;
    float s = 0.f;
    for (int nn = 0; nn < 16; ++nn) s += lds_red[wv][nn + 16 * kh][r];
    atomicAdd(&racc[b * 256 + wv * 16 + m], s);
  }
}

// h[b,:] = Wp @ (racc[b,:]/T) + bp ; out = h @ Wfc^T + bfc
__global__ __launch_bounds__(256) void wavenet_head_kernel(
    const float* __restrict__ racc, const float* __restrict__ Wp,
    const float* __restrict__ bp, const float* __restrict__ Wfc,
    const float* __restrict__ bfc, float* __restrict__ out) {
  __shared__ float h[BATCH][256];
  const int e = threadIdx.x;
  for (int b = 0; b < BATCH; ++b) {
    float acc = bp[e];
    for (int k = 0; k < 256; ++k)
      acc += Wp[e * 256 + k] * (racc[b * 256 + k] * (1.f / (float)TLEN));
    h[b][e] = acc;
  }
  __syncthreads();
  if (e < BATCH * 2) {
    const int b = e >> 1, cls = e & 1;
    float acc = bfc[cls];
    for (int k = 0; k < 256; ++k) acc += Wfc[cls * 256 + k] * h[b][k];
    out[b * 2 + cls] = acc;
  }
}

extern "C" void kernel_launch(void* const* d_in, const int* in_sizes, int n_in,
                              void* d_out, int out_size, void* d_ws, size_t ws_size,
                              hipStream_t stream) {
  const float* x   = (const float*)d_in[0];
  const float* Wc0 = (const float*)d_in[1];
  const float* bc0 = (const float*)d_in[2];
  const float* Wc  = (const float*)d_in[3];
  const float* bc  = (const float*)d_in[4];
  const float* Wr  = (const float*)d_in[5];
  const float* br  = (const float*)d_in[6];
  const float* Ws  = (const float*)d_in[7];
  const float* bs  = (const float*)d_in[8];
  const float* Wp  = (const float*)d_in[9];
  const float* bp  = (const float*)d_in[10];
  const float* Wfc = (const float*)d_in[11];
  const float* bfc = (const float*)d_in[12];

  // Workspace carve-up: x ping-pong (f32), racc (f32), packed Ws + y (bf16).
  char* wsb = (char*)d_ws;
  float* xA   = (float*)wsb;                       // [8,32,16384] f32
  float* xB   = xA + (size_t)BATCH * 32 * TLEN;
  float* racc = xB + (size_t)BATCH * 32 * TLEN;    // [8,256] f32
  __bf16* wspk = (__bf16*)(racc + BATCH * 256);    // [28][16][32][16] bf16
  __bf16* ypk  = wspk + (size_t)NLAYER * 16 * 32 * 16;  // [28][8][1024][32][16] bf16

  static const int dil[NLAYER] = {1, 2, 4, 8, 16, 32, 64, 128, 256, 512,
                                     2, 4, 8, 16, 32, 64, 128, 256, 512,
                                     2, 4, 8, 16, 32, 64, 128, 256, 512};

  wavenet_prepack_kernel<<<(NLAYER * 16 * 32 * 16 + 255) / 256, 256, 0, stream>>>(
      Ws, racc, wspk);

  dim3 grid(TLEN / 256, BATCH), blk(128);
  const float* xin = x;
  float* xout = xA;
  for (int i = 0; i < NLAYER; ++i) {
    const float* wc  = (i == 0) ? Wc0 : (Wc + (size_t)(i - 1) * 32 * 32 * 2);
    const float* bcc = (i == 0) ? bc0 : (bc + (size_t)(i - 1) * 32);
    wavenet_layer_kernel<<<grid, blk, 0, stream>>>(
        xin, xout, ypk, wc, bcc,
        Wr + (size_t)i * 32 * 32, br + (size_t)i * 32,
        dil[i], (i == 0) ? 1 : 0, i);
    xin = xout;
    xout = (xout == xA) ? xB : xA;
  }

  wavenet_skipsum_kernel<<<dim3(NTILE / 32, BATCH), 512, 0, stream>>>(
      ypk, wspk, bs, racc);
  wavenet_head_kernel<<<1, 256, 0, stream>>>(racc, Wp, bp, Wfc, bfc, (float*)d_out);
}
